// SceneGraphEncoder_46205258171003
// MI455X (gfx1250) — matrix operand order repaired and checked by reading, not compiled, not run on hardware
//
#include <hip/hip_runtime.h>
#include <hip/hip_bf16.h>

typedef __attribute__((ext_vector_type(16))) __bf16 v16bf;
typedef __attribute__((ext_vector_type(8)))  float  v8f;

#define BM 512
#define BN 64
#define BK 32
#define THREADS 256

// Direct global->LDS async copy (CDNA5 GLOBAL_LOAD_ASYNC_TO_LDS_B128).
// vdst = per-lane LDS byte address, vaddr = per-lane 64-bit global address.
// Tracked by ASYNCcnt; completion enforced with s_wait_asynccnt.
__device__ __forceinline__ void async_copy_b128(const void* lds_ptr, const void* g_ptr)
{
    const unsigned lds_addr = (unsigned)(uintptr_t)lds_ptr;        // low 32 bits = LDS offset
    const unsigned long long gaddr = (unsigned long long)(uintptr_t)g_ptr;
    asm volatile("global_load_async_to_lds_b128 %0, %1, off"
                 :: "v"(lds_addr), "v"(gaddr) : "memory");
}

__device__ __forceinline__ void wait_async0()
{
    asm volatile("s_wait_asynccnt 0x0" ::: "memory");
}

// ---------------------------------------------------------------------------
// out = relu(A[M,K](bf16) @ W[K,N] + bias), weights pre-transposed Wt[N,K] bf16.
// Block tile 512x64, 8 waves stacked over M; each wave computes a 64x64 patch
// as 4x4 WMMA fragments (16 v_wmma per K-step of 32).
// K-tiles are double-buffered in LDS and filled with async global->LDS copies
// that overlap the WMMA work of the current tile (no VGPR staging, no spills).
// A-frag (16-bit 16x32): lane<16 -> K 0..7 & 16..23 of row (lane&15);
// lane>=16 -> K 8..15 & 24..31.  B-frag: lane holds col (lane&15),
// same K order as A (from the transposed LDS tile).
// ---------------------------------------------------------------------------
__global__ __launch_bounds__(THREADS, 1) void k_gemm_bf16(
    const __bf16* __restrict__ A, const __bf16* __restrict__ Wt,
    const float* __restrict__ bias,
    __bf16* __restrict__ outH, float* __restrict__ outF,
    int M, int Nn, int K)
{
    __shared__ __bf16 As[2 * BM * BK];
    __shared__ __bf16 Bs[2 * BN * BK];

    const int tid  = threadIdx.x;
    const int lane = tid & 31;
    const int wave = tid >> 5;
    const int wm   = wave * 64;          // wave M offset inside the 512-row tile
    const int m0   = blockIdx.x * BM;
    const int n0   = blockIdx.y * BN;

    const int lrow = lane & 15;
    const int kgrp = lane >> 4;
    const int kA   = kgrp * 8;           // A fragment K sub-offset
    const int kB   = kgrp * 8;           // B fragment K sub-offset

    // async-copy work split: A tile 512x32 -> 8 x b128/thread (2 rows x 4 quads),
    // B tile 64x32 -> 1 x b128/thread.
    const int ar0 = tid * 2;             // first of two A rows this thread copies
    const int br  = tid >> 2;            // B row (0..63)
    const int bq  = (tid & 3) * 8;       // B quad offset in elements

    v8f acc[4][4] = {};

    const int nT = K / BK;

    // stage K-tile t into LDS buffer p via async copies
    auto stage = [&](int p, int k0) {
        __bf16* as = As + p * BM * BK;
        __bf16* bs = Bs + p * BN * BK;
#pragma unroll
        for (int hh = 0; hh < 2; ++hh)
#pragma unroll
            for (int q = 0; q < 4; ++q)
                async_copy_b128(as + (ar0 + hh) * BK + q * 8,
                                A + (size_t)(m0 + ar0 + hh) * K + k0 + q * 8);
        async_copy_b128(bs + br * BK + bq,
                        Wt + (size_t)(n0 + br) * K + k0 + bq);
    };

    stage(0, 0);
    wait_async0();
    __syncthreads();

    union Frag { uint4 u[2]; v16bf v; };

    for (int t = 0; t < nT; ++t) {
        const int p = t & 1;

        // kick off async copies for the next K-tile (overlaps the WMMAs below)
        if (t + 1 < nT)
            stage(p ^ 1, (t + 1) * BK);

        // compute from LDS buffer p
        {
            const __bf16* as = As + p * BM * BK;
            const __bf16* bs = Bs + p * BN * BK;
            Frag fa[4];
#pragma unroll
            for (int fi = 0; fi < 4; ++fi) {
                const __bf16* ap = as + (wm + fi * 16 + lrow) * BK;
                fa[fi].u[0] = *(const uint4*)(ap + kA);
                fa[fi].u[1] = *(const uint4*)(ap + 16 + kA);
            }
#pragma unroll
            for (int fj = 0; fj < 4; ++fj) {
                Frag fb;
                const __bf16* bp = bs + (fj * 16 + lrow) * BK + kB;
                fb.u[0] = *(const uint4*)(bp);
                fb.u[1] = *(const uint4*)(bp + 16);
#pragma unroll
                for (int fi = 0; fi < 4; ++fi) {
                    acc[fi][fj] = __builtin_amdgcn_wmma_f32_16x16x32_bf16(
                        false, fa[fi].v, false, fb.v, (short)0, acc[fi][fj],
                        false, false);
                }
            }
        }

        // make sure next tile's async copies have landed, then swap buffers
        if (t + 1 < nT)
            wait_async0();
        __syncthreads();
    }

    // Epilogue: C/D layout -> VGPR r: lanes 0-15 row r, lanes 16-31 row 8+r.
    // Uniform outF check hoisted out of the store loops (no per-element branch).
    const int rgrp = kgrp * 8;
    if (outF) {
#pragma unroll
        for (int fj = 0; fj < 4; ++fj) {
            const int col = n0 + fj * 16 + lrow;
            const float bv = bias[col];
#pragma unroll
            for (int fi = 0; fi < 4; ++fi) {
#pragma unroll
                for (int r = 0; r < 8; ++r) {
                    const int row = m0 + wm + fi * 16 + rgrp + r;
                    float v = acc[fi][fj][r] + bv;
                    v = v > 0.f ? v : 0.f;
                    const size_t o = (size_t)row * Nn + col;
                    outH[o] = (__bf16)v;
                    outF[o] = v;
                }
            }
        }
    } else {
#pragma unroll
        for (int fj = 0; fj < 4; ++fj) {
            const int col = n0 + fj * 16 + lrow;
            const float bv = bias[col];
#pragma unroll
            for (int fi = 0; fi < 4; ++fi) {
#pragma unroll
                for (int r = 0; r < 8; ++r) {
                    const int row = m0 + wm + fi * 16 + rgrp + r;
                    float v = acc[fi][fj][r] + bv;
                    v = v > 0.f ? v : 0.f;
                    outH[(size_t)row * Nn + col] = (__bf16)v;
                }
            }
        }
    }
}

// ---------------------------------------------------------------------------
__global__ void k_build_idx(const int* __restrict__ triples,
                            int* __restrict__ sIdx, int* __restrict__ oIdx,
                            int Nr, int O)
{
    int i = blockIdx.x * blockDim.x + threadIdx.x;
    if (i >= Nr) return;
    const int base = (i / O) * O;
    sIdx[i] = triples[3 * i + 0] + base;
    oIdx[i] = triples[3 * i + 2] + base;
}

// W[K,N] f32 -> Wt[N,K] bf16
__global__ void k_cvt_w(const float* __restrict__ W, __bf16* __restrict__ Wt,
                        int K, int Nn)
{
    int j = blockIdx.x * blockDim.x + threadIdx.x;
    if (j >= K * Nn) return;
    const int k = j / Nn, n = j - k * Nn;
    Wt[(size_t)n * K + k] = (__bf16)W[j];
}

// cur_t[i] = concat(ov[s[i]], pv[i], ov[o[i]]) as bf16
__global__ void k_gather(const float* __restrict__ ovF, const float* __restrict__ pvF,
                         const __bf16* __restrict__ ovH, const __bf16* __restrict__ pvH,
                         int pvStride,
                         const int* __restrict__ sIdx, const int* __restrict__ oIdx,
                         __bf16* __restrict__ out, long long total, int din, int srcF32)
{
    long long j = (long long)blockIdx.x * blockDim.x + threadIdx.x;
    if (j >= total) return;
    const int W3 = 3 * din;
    const long long i = j / W3;
    const int c = (int)(j - i * W3);
    __bf16 v;
    if (c < din) {
        const long long s = sIdx[i];
        v = srcF32 ? (__bf16)ovF[s * din + c] : ovH[s * din + c];
    } else if (c < 2 * din) {
        const int cc = c - din;
        v = srcF32 ? (__bf16)pvF[i * pvStride + cc] : pvH[i * pvStride + cc];
    } else {
        const int cc = c - 2 * din;
        const long long o = oIdx[i];
        v = srcF32 ? (__bf16)ovF[o * din + cc] : ovH[o * din + cc];
    }
    out[j] = v;
}

// pooled[s[i]] += new_t[i, 0:h];  pooled[o[i]] += new_t[i, h+dout:2h+dout]
__global__ void k_scatter(const float* __restrict__ newt,
                          const int* __restrict__ sIdx, const int* __restrict__ oIdx,
                          float* __restrict__ pooled, int Nr, int h, int W, int dout)
{
    long long j = (long long)blockIdx.x * blockDim.x + threadIdx.x;
    if (j >= (long long)Nr * h) return;
    const int i = (int)(j / h);
    const int c = (int)(j - (long long)i * h);
    atomicAdd(&pooled[(size_t)sIdx[i] * h + c], newt[(size_t)i * W + c]);
    atomicAdd(&pooled[(size_t)oIdx[i] * h + c], newt[(size_t)i * W + h + dout + c]);
}

__global__ void k_count(const int* __restrict__ sIdx, const int* __restrict__ oIdx,
                        float* __restrict__ cnt, int Nr)
{
    int i = blockIdx.x * blockDim.x + threadIdx.x;
    if (i >= Nr) return;
    atomicAdd(&cnt[sIdx[i]], 1.f);
    atomicAdd(&cnt[oIdx[i]], 1.f);
}

__global__ void k_finalize(const float* __restrict__ pooled, const float* __restrict__ cnt,
                           __bf16* __restrict__ out, int Nr, int h)
{
    long long j = (long long)blockIdx.x * blockDim.x + threadIdx.x;
    if (j >= (long long)Nr * h) return;
    const int i = (int)(j / h);
    float d = cnt[i];
    d = d > 1.f ? d : 1.f;
    out[j] = (__bf16)(pooled[j] / d);
}

// ---------------------------------------------------------------------------
extern "C" void kernel_launch(void* const* d_in, const int* in_sizes, int n_in,
                              void* d_out, int out_size, void* d_ws, size_t ws_size,
                              hipStream_t stream)
{
    (void)in_sizes; (void)n_in; (void)out_size; (void)ws_size;
    const int B = 64, O = 2048, Nr = B * O;          // 131072 rows

    const float* obj     = (const float*)d_in[0];
    const float* pred    = (const float*)d_in[1];
    const int*   triples = (const int*)d_in[2];

    struct LDims { int din, h, dout; };
    const LDims ld[6] = {{128,128,256},{256,256,256},{256,256,256},
                         {256,256,256},{256,256,256},{256,256,256}};

    // Workspace carving (256B aligned)
    char* ws = (char*)d_ws;
    size_t off = 0;
    auto carve = [&](size_t bytes) -> void* {
        void* p = ws + off;
        off = (off + bytes + 255) & ~(size_t)255;
        return p;
    };
    int*    sIdx   = (int*)   carve((size_t)Nr * 4);
    int*    oIdx   = (int*)   carve((size_t)Nr * 4);
    __bf16* bufA   = (__bf16*)carve((size_t)Nr * 768 * 2);   // cur_t / h2
    __bf16* bufB   = (__bf16*)carve((size_t)Nr * 256 * 2);   // h1 / pooled-bf16
    __bf16* bufC   = (__bf16*)carve((size_t)Nr * 768 * 2);   // new_t bf16 (pv source)
    float*  newtF  = (float*) carve((size_t)Nr * 768 * 4);   // new_t f32 (scatter src)
    float*  pooled = (float*) carve((size_t)Nr * 256 * 4);
    float*  cnt    = (float*) carve((size_t)Nr * 4);
    __bf16* ovb    = (__bf16*)carve((size_t)Nr * 256 * 2);   // new ov bf16
    __bf16* wt1b   = (__bf16*)carve((size_t)768 * 256 * 2);
    __bf16* wt2b   = (__bf16*)carve((size_t)256 * 768 * 2);
    __bf16* wt3b   = (__bf16*)carve((size_t)256 * 256 * 2);
    __bf16* wt4b   = (__bf16*)carve((size_t)256 * 256 * 2);

    k_build_idx<<<(Nr + 255) / 256, 256, 0, stream>>>(triples, sIdx, oIdx, Nr, O);

    int h_prev = 0, stride_prev = 0;
    for (int L = 0; L < 6; ++L) {
        const int din = ld[L].din, h = ld[L].h, dout = ld[L].dout;
        const int K1 = 3 * din, N1 = h;
        const int K2 = h,       N2 = 2 * h + dout;
        const float* W1 = (const float*)d_in[3 + 8 * L + 0];
        const float* b1 = (const float*)d_in[3 + 8 * L + 1];
        const float* W2 = (const float*)d_in[3 + 8 * L + 2];
        const float* b2 = (const float*)d_in[3 + 8 * L + 3];
        const float* W3 = (const float*)d_in[3 + 8 * L + 4];
        const float* b3 = (const float*)d_in[3 + 8 * L + 5];
        const float* W4 = (const float*)d_in[3 + 8 * L + 6];
        const float* b4 = (const float*)d_in[3 + 8 * L + 7];

        // weight convert + transpose to [N,K] bf16
        k_cvt_w<<<(K1 * N1 + 255) / 256, 256, 0, stream>>>(W1, wt1b, K1, N1);
        k_cvt_w<<<(K2 * N2 + 255) / 256, 256, 0, stream>>>(W2, wt2b, K2, N2);
        k_cvt_w<<<(h * h + 255) / 256, 256, 0, stream>>>(W3, wt3b, h, h);
        k_cvt_w<<<(h * dout + 255) / 256, 256, 0, stream>>>(W4, wt4b, h, dout);

        // gather -> cur_t (bf16)
        const long long tot = (long long)Nr * K1;
        const int gblk = (int)((tot + 255) / 256);
        if (L == 0)
            k_gather<<<gblk, 256, 0, stream>>>(obj, pred, nullptr, nullptr, din,
                                               sIdx, oIdx, bufA, tot, din, 1);
        else
            k_gather<<<gblk, 256, 0, stream>>>(nullptr, nullptr, ovb, bufC + h_prev,
                                               stride_prev, sIdx, oIdx, bufA, tot, din, 0);

        // net1: two WMMA GEMMs
        k_gemm_bf16<<<dim3(Nr / BM, N1 / BN), THREADS, 0, stream>>>(
            bufA, wt1b, b1, bufB, nullptr, Nr, N1, K1);
        k_gemm_bf16<<<dim3(Nr / BM, N2 / BN), THREADS, 0, stream>>>(
            bufB, wt2b, b2, bufC, newtF, Nr, N2, K2);

        // scatter-average pooling
        hipMemsetAsync(pooled, 0, (size_t)Nr * h * 4, stream);
        hipMemsetAsync(cnt, 0, (size_t)Nr * 4, stream);
        k_scatter<<<(int)(((long long)Nr * h + 255) / 256), 256, 0, stream>>>(
            newtF, sIdx, oIdx, pooled, Nr, h, N2, dout);
        k_count<<<(Nr + 255) / 256, 256, 0, stream>>>(sIdx, oIdx, cnt, Nr);
        k_finalize<<<(int)(((long long)Nr * h + 255) / 256), 256, 0, stream>>>(
            pooled, cnt, bufB, Nr, h);

        // net2: two WMMA GEMMs; final layer also writes f32 to d_out
        k_gemm_bf16<<<dim3(Nr / BM, h / BN), THREADS, 0, stream>>>(
            bufB, wt3b, b3, bufA, nullptr, Nr, h, h);
        float* of = (L == 5) ? (float*)d_out : nullptr;
        k_gemm_bf16<<<dim3(Nr / BM, dout / BN), THREADS, 0, stream>>>(
            bufA, wt4b, b4, ovb, of, Nr, dout, h);

        h_prev = h;
        stride_prev = N2;
    }
}
